// DescrptDPA3V7_22986664968684
// MI455X (gfx1250) — matrix operand order repaired and checked by reading, not compiled, not run on hardware
//
#include <hip/hip_runtime.h>
#include <stdint.h>

typedef unsigned short u16;
typedef unsigned int   u32;
typedef __attribute__((ext_vector_type(16))) __bf16 v16bf;
typedef __attribute__((ext_vector_type(8)))  float  v8f;

#define N_LOCc   1024
#define N_DIMc   128
#define E_DIMc   64
#define A_DIMc   32
#define HIDc     128
#define N_EDGEc  65536
#define N_ANGLEc 262144
#define NBASISc  7

// ---------------------------------------------------------------------------
// helpers
// ---------------------------------------------------------------------------
__device__ __forceinline__ u16 f2bf(float f) {
  u32 u = __float_as_uint(f);
  u32 r = (u + 0x7FFFu + ((u >> 16) & 1u)) >> 16;
  return (u16)r;
}

// order-preserving float<->uint encoding for atomic max
__device__ __forceinline__ u32 encf(float f) {
  u32 u = __float_as_uint(f);
  return (u & 0x80000000u) ? ~u : (u | 0x80000000u);
}
__device__ __forceinline__ float decf(u32 e) {
  u32 u = (e & 0x80000000u) ? (e & 0x7FFFFFFFu) : ~e;
  return __uint_as_float(u);
}

__device__ __forceinline__ float siluf(float x) { return x / (1.0f + __expf(-x)); }
__device__ __forceinline__ float sigmf(float x) { return 1.0f / (1.0f + __expf(-x)); }

union FragU { uint4 q[2]; v16bf v; };

// A fragment (16x32 bf16, M x K): lanes 0-15 row M=lane, K = {0..7, 16..23};
// lanes 16-31 row M=lane-16, K = {8..15, 24..31}.  base points at (m0, kt*32).
__device__ __forceinline__ v16bf load_a_frag(const u16* base, int pitch, int lane) {
  int r = lane & 15;
  int koff = (lane & 16) ? 8 : 0;
  const u16* p = base + (size_t)r * pitch + koff;
  FragU f;
  f.q[0] = *(const uint4*)(p);
  f.q[1] = *(const uint4*)(p + 16);
  return f.v;
}

// B fragment (32x16 bf16, K x N) stored as Wt[N,K] row-major:
// lanes 0-15: col N=lane, K = {0..7, 16..23}; lanes 16-31: K = {8..15, 24..31}.
__device__ __forceinline__ v16bf load_b_frag(const u16* base, int pitchK, int lane) {
  int n = lane & 15;
  int koff = (lane & 16) ? 8 : 0;
  const u16* p = base + (size_t)n * pitchK + koff;
  FragU f;
  f.q[0] = *(const uint4*)(p);
  f.q[1] = *(const uint4*)(p + 16);
  return f.v;
}

#define WMMA_BF16(a, b, c) \
  __builtin_amdgcn_wmma_f32_16x16x32_bf16(false, (a), false, (b), (short)0, (c), false, false)

// ---------------------------------------------------------------------------
// WMMA linear: out[M,N] = A[M,K](bf16, pitch lda) @ W[K,N] (given as Wt[N,K]) + bias
// block = 256 threads = 8 waves; wave handles a 16-row slice.
// M multiple of 128, K multiple of 32, N multiple of 16.
// ---------------------------------------------------------------------------
__global__ void k_wmma_lin(const u16* __restrict__ A, int lda,
                           const u16* __restrict__ Wt, const float* __restrict__ bias,
                           float* __restrict__ out, int M, int K, int N) {
  int lane = threadIdx.x & 31;
  int wave = (blockIdx.x * blockDim.x + threadIdx.x) >> 5;
  int m0 = wave << 4;
  if (m0 >= M) return;
  for (int nt = 0; nt < N; nt += 16) {
    v8f acc = {0, 0, 0, 0, 0, 0, 0, 0};
    for (int kt = 0; kt < K; kt += 32) {
      v16bf a = load_a_frag(A + (size_t)m0 * lda + kt, lda, lane);
      v16bf b = load_b_frag(Wt + (size_t)nt * K + kt, K, lane);
      acc = WMMA_BF16(a, b, acc);
    }
    int n = nt + (lane & 15);
    float bv = bias ? bias[n] : 0.0f;
    int mb = m0 + ((lane >> 4) << 3);   // D layout: lanes>=16 hold M = r+8
#pragma unroll
    for (int r2 = 0; r2 < 8; ++r2)
      out[(size_t)(mb + r2) * N + n] = acc[r2] + bv;
  }
}

// ---------------------------------------------------------------------------
// Fused gated MLP: out = (silu(X@Wg + bg) * (X@Wu + bu)) @ Wd + bd
// X[M,K] bf16; weights pre-transposed [N,K] bf16; HID fixed 128.
// block = 128 threads = 4 waves; wave handles 16 rows; h staged bf16 in LDS.
// ---------------------------------------------------------------------------
#define GW 4
__global__ void k_wmma_gmlp(const u16* __restrict__ X, int K,
                            const u16* __restrict__ Wgt, const float* __restrict__ bg,
                            const u16* __restrict__ Wut, const float* __restrict__ bu,
                            const u16* __restrict__ Wdt, const float* __restrict__ bd,
                            float* __restrict__ out, int M, int NOUT) {
  __shared__ __align__(16) u16 lds[GW * 16 * 136];
  int lane = threadIdx.x & 31;
  int widx = threadIdx.x >> 5;
  int wave = blockIdx.x * GW + widx;
  int m0 = wave << 4;
  u16* h = lds + widx * (16 * 136);
  bool active = (m0 < M);
  if (active) {
    for (int ht = 0; ht < HIDc; ht += 16) {
      v8f ag = {0, 0, 0, 0, 0, 0, 0, 0};
      v8f au = {0, 0, 0, 0, 0, 0, 0, 0};
      for (int kt = 0; kt < K; kt += 32) {
        v16bf a   = load_a_frag(X + (size_t)m0 * K + kt, K, lane);
        v16bf bgf = load_b_frag(Wgt + (size_t)ht * K + kt, K, lane);
        v16bf buf = load_b_frag(Wut + (size_t)ht * K + kt, K, lane);
        ag = WMMA_BF16(a, bgf, ag);
        au = WMMA_BF16(a, buf, au);
      }
      int n = ht + (lane & 15);
      float bgv = bg[n], buv = bu[n];
      int mb = (lane >> 4) << 3;
#pragma unroll
      for (int r2 = 0; r2 < 8; ++r2) {
        float g = ag[r2] + bgv;
        float uu = au[r2] + buv;
        h[(mb + r2) * 136 + n] = f2bf(siluf(g) * uu);
      }
    }
  }
  __syncthreads();
  if (active) {
    for (int ot = 0; ot < NOUT; ot += 16) {
      v8f acc = {0, 0, 0, 0, 0, 0, 0, 0};
      for (int kt = 0; kt < HIDc; kt += 32) {
        v16bf a = load_a_frag(h + kt, 136, lane);
        v16bf b = load_b_frag(Wdt + (size_t)ot * HIDc + kt, HIDc, lane);
        acc = WMMA_BF16(a, b, acc);
      }
      int n = ot + (lane & 15);
      float bv = bd[n];
      int mb = m0 + ((lane >> 4) << 3);
#pragma unroll
      for (int r2 = 0; r2 < 8; ++r2)
        out[(size_t)(mb + r2) * NOUT + n] = acc[r2] + bv;
    }
  }
}

// ---------------------------------------------------------------------------
// elementwise / gather / scatter kernels
// ---------------------------------------------------------------------------
__global__ void k_conv_t(const float* __restrict__ W, u16* __restrict__ Wt, int K, int N) {
  int i = blockIdx.x * blockDim.x + threadIdx.x;
  if (i >= K * N) return;
  int k = i / N, n = i - k * N;
  Wt[(size_t)n * K + k] = f2bf(W[i]);
}

__global__ void k_conv_scaled(const float* __restrict__ x, u16* __restrict__ y, float s, int n) {
  int i = blockIdx.x * blockDim.x + threadIdx.x;
  if (i >= n) return;
  y[i] = f2bf(x[i] * s);
}

// line_info = [angle(32) | edge[eik](64) | edge[eij](64)] -> bf16
__global__ void k_concat_line(const float* __restrict__ ang, const float* __restrict__ edge,
                              const int* __restrict__ eik, const int* __restrict__ eij,
                              u16* __restrict__ dst) {
  const int C = A_DIMc + 2 * E_DIMc;  // 160
  int i = blockIdx.x * blockDim.x + threadIdx.x;
  if (i >= N_ANGLEc * C) return;
  int row = i / C, c = i - row * C;
  float v;
  if (c < A_DIMc)               v = ang[(size_t)row * A_DIMc + c];
  else if (c < A_DIMc + E_DIMc) v = edge[(size_t)eik[row] * E_DIMc + (c - A_DIMc)];
  else                          v = edge[(size_t)eij[row] * E_DIMc + (c - A_DIMc - E_DIMc)];
  dst[i] = f2bf(v);
}

// edge_info = [node[n2e](128) | node_ext[next2e](128) | edge(64)] -> bf16
__global__ void k_concat_edge(const float* __restrict__ node, const float* __restrict__ node_ext,
                              const float* __restrict__ edge,
                              const int* __restrict__ n2e, const int* __restrict__ nx2e,
                              u16* __restrict__ dst) {
  const int C = 2 * N_DIMc + E_DIMc;  // 320
  int i = blockIdx.x * blockDim.x + threadIdx.x;
  if (i >= N_EDGEc * C) return;
  int row = i / C, c = i - row * C;
  float v;
  if (c < N_DIMc)              v = node[(size_t)n2e[row] * N_DIMc + c];
  else if (c < 2 * N_DIMc)     v = node_ext[(size_t)nx2e[row] * N_DIMc + (c - N_DIMc)];
  else                         v = edge[(size_t)row * E_DIMc + (c - 2 * N_DIMc)];
  dst[i] = f2bf(v);
}

// node_info = [node(128) | agg(64)*scale] -> bf16
__global__ void k_concat_node(const float* __restrict__ node, const float* __restrict__ agg,
                              float scale, u16* __restrict__ dst) {
  const int C = N_DIMc + E_DIMc;  // 192
  int i = blockIdx.x * blockDim.x + threadIdx.x;
  if (i >= N_LOCc * C) return;
  int row = i / C, c = i - row * C;
  float v = (c < N_DIMc) ? node[(size_t)row * N_DIMc + c]
                         : agg[(size_t)row * E_DIMc + (c - N_DIMc)] * scale;
  dst[i] = f2bf(v);
}

// refine_line_info = [angle1(32) | node1[n2a](128) | e2[eik](64) | e2[eij](64)] -> bf16
__global__ void k_concat_rline(const float* __restrict__ ang1, const float* __restrict__ node1,
                               const float* __restrict__ e2,
                               const int* __restrict__ n2a, const int* __restrict__ eik,
                               const int* __restrict__ eij, u16* __restrict__ dst) {
  const int C = A_DIMc + N_DIMc + 2 * E_DIMc;  // 288
  int i = blockIdx.x * blockDim.x + threadIdx.x;
  if (i >= N_ANGLEc * C) return;
  int row = i / C, c = i - row * C;
  float v;
  if (c < A_DIMc)                        v = ang1[(size_t)row * A_DIMc + c];
  else if (c < A_DIMc + N_DIMc)          v = node1[(size_t)n2a[row] * N_DIMc + (c - A_DIMc)];
  else if (c < A_DIMc + N_DIMc + E_DIMc) v = e2[(size_t)eik[row] * E_DIMc + (c - A_DIMc - N_DIMc)];
  else                                   v = e2[(size_t)eij[row] * E_DIMc + (c - A_DIMc - N_DIMc - E_DIMc)];
  dst[i] = f2bf(v);
}

// ---- dimwise segment softmax (logits scaled by sw), then weighted aggregate ----
__global__ void k_seg_max(const float* __restrict__ lg, const float* __restrict__ swv,
                          const int* __restrict__ seg, int rows, int dim, u32* __restrict__ smax) {
  int i = blockIdx.x * blockDim.x + threadIdx.x;
  if (i >= rows * dim) return;
  int row = i / dim, d = i - row * dim;
  atomicMax(&smax[(size_t)seg[row] * dim + d], encf(lg[i] * swv[row]));
}

__global__ void k_seg_expsum(const float* __restrict__ lg, const float* __restrict__ swv,
                             const int* __restrict__ seg, int rows, int dim,
                             const u32* __restrict__ smax, float* __restrict__ ssum) {
  int i = blockIdx.x * blockDim.x + threadIdx.x;
  if (i >= rows * dim) return;
  int row = i / dim, d = i - row * dim;
  int s = seg[row];
  float m = decf(smax[(size_t)s * dim + d]);
  if (!(fabsf(m) <= 3.0e38f)) m = 0.0f;  // non-finite -> 0 (matches reference)
  atomicAdd(&ssum[(size_t)s * dim + d], __expf(lg[i] * swv[row] - m));
}

__global__ void k_msg_agg(const float* __restrict__ lg, const float* __restrict__ swv,
                          const int* __restrict__ seg, const u32* __restrict__ smax,
                          const float* __restrict__ ssum, const float* __restrict__ val,
                          int rows, int dim, float* __restrict__ agg) {
  int i = blockIdx.x * blockDim.x + threadIdx.x;
  if (i >= rows * dim) return;
  int row = i / dim, d = i - row * dim;
  int s = seg[row];
  float m = decf(smax[(size_t)s * dim + d]);
  if (!(fabsf(m) <= 3.0e38f)) m = 0.0f;
  float e = __expf(lg[i] * swv[row] - m);
  float alpha = e / fmaxf(ssum[(size_t)s * dim + d], 1e-30f);
  atomicAdd(&agg[(size_t)s * dim + d], alpha * val[i] * swv[row]);
}

// sm = feat * sigmoid(rbf @ Wenv) * sw ; atomic aggregate over seg
__global__ void k_env_sm_agg(const float* __restrict__ feat, const float* __restrict__ rbf,
                             const float* __restrict__ Wenv, const float* __restrict__ swv,
                             const int* __restrict__ seg, int rows, int dim,
                             float* __restrict__ agg) {
  int i = blockIdx.x * blockDim.x + threadIdx.x;
  if (i >= rows * dim) return;
  int row = i / dim, d = i - row * dim;
  float z = 0.0f;
#pragma unroll
  for (int k = 0; k < NBASISc; ++k) z += rbf[(size_t)row * NBASISc + k] * Wenv[k * dim + d];
  atomicAdd(&agg[(size_t)seg[row] * dim + d], feat[i] * sigmf(z) * swv[row]);
}

// out = f(a*ascale) + res[d] * b   (f = silu or identity)
__global__ void k_scale_res(const float* __restrict__ a, float ascale, int dosilu,
                            const float* __restrict__ res, const float* __restrict__ b,
                            float* __restrict__ out, int rows, int dim) {
  int i = blockIdx.x * blockDim.x + threadIdx.x;
  if (i >= rows * dim) return;
  int d = i % dim;
  float x = a[i] * ascale;
  if (dosilu) x = siluf(x);
  out[i] = x + res[d] * b[i];
}

// ---------------------------------------------------------------------------
// host orchestration
// ---------------------------------------------------------------------------
extern "C" void kernel_launch(void* const* d_in, const int* in_sizes, int n_in,
                              void* d_out, int out_size, void* d_ws, size_t ws_size,
                              hipStream_t stream) {
  (void)in_sizes; (void)n_in; (void)out_size; (void)ws_size;

  const float* node_ext = (const float*)d_in[0];   // [1024,128] (NLOC==NALL -> also node_ebd)
  const float* edge0    = (const float*)d_in[1];   // [65536,64]
  const float* ang0     = (const float*)d_in[3];   // [262144,32]
  const float* swv      = (const float*)d_in[6];   // [65536]
  const float* a_swv    = (const float*)d_in[9];   // [262144]
  const int*   eidx     = (const int*)d_in[10];    // [2,65536]
  const int*   aidx     = (const int*)d_in[11];    // [3,262144]
  const float* erbf     = (const float*)d_in[12];  // [65536,7]
  const float* arbf     = (const float*)d_in[13];  // [262144,7]
  const int* n2e = eidx;
  const int* nx2e = eidx + N_EDGEc;
  const int* n2a = aidx;
  const int* eij2a = aidx + N_ANGLEc;
  const int* eik2a = aidx + 2 * N_ANGLEc;

  // params flattened in recursive dict-insertion order (gmlp: wg,bg,wu,bu,wd,bd; lin: w,b)
  //  0-5  line_attn_edge_mlp     6 lawl.w      7-8  line_attn_angle_mlp(w,b)
  //  9-14 atom_attn_edge_mlp    15 aawl.w     16-21 atom_attn_node_mlp
  // 22-27 line_refine_mlp       28 lre.w      29-30 lrnf(w,b)  31-32 lref(w,b)
  // 33-38 atom_refine_mlp       39 are.w      40-41 arnf(w,b)  42-43 aref(w,b)
  // 44 lae_res 45 laa_res 46 aan_res 47 aae_res 48 lre_res 49 lra_res 50 arn_res 51 are_res
  const float* P[52];
  for (int i = 0; i < 52; ++i) P[i] = (const float*)d_in[14 + i];

  char* ws = (char*)d_ws;
  size_t off = 0;
  auto alloc = [&](size_t bytes) -> void* {
    void* p = ws + off;
    off = (off + bytes + 255) & ~(size_t)255;
    return p;
  };
  auto CT = [&](const float* W, int K, int N) -> u16* {  // convert + transpose weight -> [N,K] bf16
    u16* t = (u16*)alloc((size_t)K * N * 2);
    int n = K * N;
    k_conv_t<<<(n + 255) / 256, 256, 0, stream>>>(W, t, K, N);
    return t;
  };

  // ---- weights (bf16, transposed) ----
  u16* lae_wg = CT(P[0], 160, 128);  u16* lae_wu = CT(P[2], 160, 128);  u16* lae_wd = CT(P[4], 128, 64);
  u16* lawl   = CT(P[6], 32, 64);
  u16* laam   = CT(P[7], 160, 32);
  u16* aae_wg = CT(P[9], 320, 128);  u16* aae_wu = CT(P[11], 320, 128); u16* aae_wd = CT(P[13], 128, 64);
  u16* aawl   = CT(P[15], 64, 64);
  u16* aan_wg = CT(P[16], 192, 128); u16* aan_wu = CT(P[18], 192, 128); u16* aan_wd = CT(P[20], 128, 128);
  u16* lrm_wg = CT(P[22], 288, 128); u16* lrm_wu = CT(P[24], 288, 128); u16* lrm_wd = CT(P[26], 128, 64);
  u16* lrnf   = CT(P[29], 64, 64);
  u16* lref   = CT(P[31], 64, 32);
  u16* arm_wg = CT(P[33], 320, 128); u16* arm_wu = CT(P[35], 320, 128); u16* arm_wd = CT(P[37], 128, 64);
  u16* arnf   = CT(P[40], 64, 128);
  u16* aref   = CT(P[42], 64, 64);

  // ---- workspace buffers ----
  u16*   BUF_A  = (u16*)alloc((size_t)N_ANGLEc * 288 * 2);      // line_info / refine_line_info / edge_info
  float* BUF_B  = (float*)alloc((size_t)N_ANGLEc * 64 * 4);     // gmlp out / angle raws
  float* BUF_L  = (float*)alloc((size_t)N_ANGLEc * 64 * 4);     // logits / refine_line_feat
  float* ANG1   = (float*)alloc((size_t)N_ANGLEc * 32 * 4);
  u16*   RLF_BF = (u16*)alloc((size_t)N_ANGLEc * 64 * 2);
  u32*   SMAX   = (u32*)alloc((size_t)N_EDGEc * 64 * 4);
  float* SSUM   = (float*)alloc((size_t)N_EDGEc * 64 * 4);      // also reused as edge-sized f32 temp
  float* AGG    = (float*)alloc((size_t)N_EDGEc * 64 * 4);      // line agg / line_refine agg
  float* E1     = (float*)alloc((size_t)N_EDGEc * 64 * 4);
  float* E2     = (float*)alloc((size_t)N_EDGEc * 64 * 4);
  float* E3     = (float*)alloc((size_t)N_EDGEc * 64 * 4);
  float* FEAT_E = (float*)alloc((size_t)N_EDGEc * 64 * 4);      // atom_edge_update / refine_atom_feat
  u16*   EBF    = (u16*)alloc((size_t)N_EDGEc * 64 * 2);        // bf16 edge-sized temp
  u32*   SM2    = (u32*)alloc((size_t)N_LOCc * 64 * 4);
  float* SS2    = (float*)alloc((size_t)N_LOCc * 64 * 4);
  float* NAG    = (float*)alloc((size_t)N_LOCc * 64 * 4);
  float* ARA    = (float*)alloc((size_t)N_LOCc * 64 * 4);
  u16*   ABF64  = (u16*)alloc((size_t)N_LOCc * 64 * 2);
  u16*   NIBF   = (u16*)alloc((size_t)N_LOCc * 192 * 2);
  float* NODE1  = (float*)alloc((size_t)N_LOCc * 128 * 4);
  float* NTMP   = (float*)alloc((size_t)N_LOCc * 128 * 4);

  float* outN = (float*)d_out;
  float* outE = outN + (size_t)N_LOCc * N_DIMc;
  float* outA = outE + (size_t)N_EDGEc * E_DIMc;

  const float invSqrtA = 0.7905694150420949f;  // 1/sqrt(16/10)
  const float invE     = 0.15625f;             // 1/(64/10)

  auto g1 = [](long n) { return dim3((unsigned)((n + 255) / 256)); };
  dim3 b256(256), b128(128);
  dim3 linA(N_ANGLEc / 128), linE(N_EDGEc / 128), linN(N_LOCc / 128);
  dim3 gmA(N_ANGLEc / 64), gmE(N_EDGEc / 64), gmN(N_LOCc / 64);

  // ================= phase 1: line attention =================
  hipMemsetAsync(SMAX, 0, (size_t)N_EDGEc * 64 * 4, stream);
  hipMemsetAsync(SSUM, 0, (size_t)N_EDGEc * 64 * 4, stream);
  hipMemsetAsync(AGG, 0, (size_t)N_EDGEc * 64 * 4, stream);

  k_concat_line<<<g1((long)N_ANGLEc * 160), b256, 0, stream>>>(ang0, edge0, eik2a, eij2a, BUF_A);
  // logits = angle_ebd @ Wwl  (angle cols 0..31 of line_info, lda=160)
  k_wmma_lin<<<linA, b256, 0, stream>>>(BUF_A, 160, lawl, nullptr, BUF_L, N_ANGLEc, 32, 64);
  k_seg_max<<<g1((long)N_ANGLEc * 64), b256, 0, stream>>>(BUF_L, a_swv, eij2a, N_ANGLEc, 64, SMAX);
  k_seg_expsum<<<g1((long)N_ANGLEc * 64), b256, 0, stream>>>(BUF_L, a_swv, eij2a, N_ANGLEc, 64, SMAX, SSUM);
  k_wmma_gmlp<<<gmA, b128, 0, stream>>>(BUF_A, 160, lae_wg, P[1], lae_wu, P[3], lae_wd, P[5],
                                        BUF_B, N_ANGLEc, 64);
  k_msg_agg<<<g1((long)N_ANGLEc * 64), b256, 0, stream>>>(BUF_L, a_swv, eij2a, SMAX, SSUM, BUF_B,
                                                          N_ANGLEc, 64, AGG);
  k_wmma_lin<<<linA, b256, 0, stream>>>(BUF_A, 160, laam, P[8], BUF_B, N_ANGLEc, 160, 32);
  k_scale_res<<<g1((long)N_ANGLEc * 32), b256, 0, stream>>>(BUF_B, 1.0f, 1, P[45], ang0, ANG1, N_ANGLEc, 32);
  k_scale_res<<<g1((long)N_EDGEc * 64), b256, 0, stream>>>(AGG, invSqrtA, 0, P[44], edge0, E1, N_EDGEc, 64);

  // ================= phase 2: atom attention =================
  hipMemsetAsync(SM2, 0, (size_t)N_LOCc * 64 * 4, stream);
  hipMemsetAsync(SS2, 0, (size_t)N_LOCc * 64 * 4, stream);
  hipMemsetAsync(NAG, 0, (size_t)N_LOCc * 64 * 4, stream);

  k_concat_edge<<<g1((long)N_EDGEc * 320), b256, 0, stream>>>(node_ext, node_ext, E1, n2e, nx2e, BUF_A);
  k_wmma_gmlp<<<gmE, b128, 0, stream>>>(BUF_A, 320, aae_wg, P[10], aae_wu, P[12], aae_wd, P[14],
                                        FEAT_E, N_EDGEc, 64);
  // atom logits from edge1 bf16 slice (cols 256..319 of edge_info, lda=320)
  k_wmma_lin<<<linE, b256, 0, stream>>>(BUF_A + 256, 320, aawl, nullptr, BUF_L, N_EDGEc, 64, 64);
  k_seg_max<<<g1((long)N_EDGEc * 64), b256, 0, stream>>>(BUF_L, swv, n2e, N_EDGEc, 64, SM2);
  k_seg_expsum<<<g1((long)N_EDGEc * 64), b256, 0, stream>>>(BUF_L, swv, n2e, N_EDGEc, 64, SM2, SS2);
  k_msg_agg<<<g1((long)N_EDGEc * 64), b256, 0, stream>>>(BUF_L, swv, n2e, SM2, SS2, FEAT_E,
                                                         N_EDGEc, 64, NAG);
  k_concat_node<<<g1((long)N_LOCc * 192), b256, 0, stream>>>(node_ext, NAG, invE, NIBF);
  k_wmma_gmlp<<<gmN, b128, 0, stream>>>(NIBF, 192, aan_wg, P[17], aan_wu, P[19], aan_wd, P[21],
                                        NTMP, N_LOCc, 128);
  k_scale_res<<<g1((long)N_LOCc * 128), b256, 0, stream>>>(NTMP, 1.0f, 0, P[46], node_ext, NODE1, N_LOCc, 128);
  k_scale_res<<<g1((long)N_EDGEc * 64), b256, 0, stream>>>(FEAT_E, 1.0f, 0, P[47], E1, E2, N_EDGEc, 64);

  // ================= phase 3: line refine =================
  hipMemsetAsync(AGG, 0, (size_t)N_EDGEc * 64 * 4, stream);

  k_concat_rline<<<g1((long)N_ANGLEc * 288), b256, 0, stream>>>(ANG1, NODE1, E2, n2a, eik2a, eij2a, BUF_A);
  k_wmma_gmlp<<<gmA, b128, 0, stream>>>(BUF_A, 288, lrm_wg, P[23], lrm_wu, P[25], lrm_wd, P[27],
                                        BUF_L, N_ANGLEc, 64);
  k_conv_scaled<<<g1((long)N_ANGLEc * 64), b256, 0, stream>>>(BUF_L, RLF_BF, 1.0f, N_ANGLEc * 64);
  k_env_sm_agg<<<g1((long)N_ANGLEc * 64), b256, 0, stream>>>(BUF_L, arbf, P[28], a_swv, eij2a,
                                                             N_ANGLEc, 64, AGG);
  k_conv_scaled<<<g1((long)N_EDGEc * 64), b256, 0, stream>>>(AGG, EBF, invSqrtA, N_EDGEc * 64);
  k_wmma_lin<<<linE, b256, 0, stream>>>(EBF, 64, lrnf, P[30], SSUM, N_EDGEc, 64, 64);
  k_scale_res<<<g1((long)N_EDGEc * 64), b256, 0, stream>>>(SSUM, 1.0f, 1, P[48], E2, E3, N_EDGEc, 64);
  k_wmma_lin<<<linA, b256, 0, stream>>>(RLF_BF, 64, lref, P[32], BUF_B, N_ANGLEc, 64, 32);
  k_scale_res<<<g1((long)N_ANGLEc * 32), b256, 0, stream>>>(BUF_B, 1.0f, 1, P[49], ANG1, outA, N_ANGLEc, 32);

  // ================= phase 4: atom refine =================
  hipMemsetAsync(ARA, 0, (size_t)N_LOCc * 64 * 4, stream);

  k_concat_edge<<<g1((long)N_EDGEc * 320), b256, 0, stream>>>(NODE1, node_ext, E3, n2e, nx2e, BUF_A);
  k_wmma_gmlp<<<gmE, b128, 0, stream>>>(BUF_A, 320, arm_wg, P[34], arm_wu, P[36], arm_wd, P[38],
                                        FEAT_E, N_EDGEc, 64);
  k_conv_scaled<<<g1((long)N_EDGEc * 64), b256, 0, stream>>>(FEAT_E, EBF, 1.0f, N_EDGEc * 64);
  k_env_sm_agg<<<g1((long)N_EDGEc * 64), b256, 0, stream>>>(FEAT_E, erbf, P[39], swv, n2e,
                                                            N_EDGEc, 64, ARA);
  k_conv_scaled<<<g1((long)N_LOCc * 64), b256, 0, stream>>>(ARA, ABF64, invE, N_LOCc * 64);
  k_wmma_lin<<<linN, b256, 0, stream>>>(ABF64, 64, arnf, P[41], NTMP, N_LOCc, 64, 128);
  k_scale_res<<<g1((long)N_LOCc * 128), b256, 0, stream>>>(NTMP, 1.0f, 1, P[50], NODE1, outN, N_LOCc, 128);
  k_wmma_lin<<<linE, b256, 0, stream>>>(EBF, 64, aref, P[43], SSUM, N_EDGEc, 64, 64);
  k_scale_res<<<g1((long)N_EDGEc * 64), b256, 0, stream>>>(SSUM, 1.0f, 1, P[51], E3, outE, N_EDGEc, 64);
}
